// DARNN_47485158424750
// MI455X (gfx1250) — matrix operand order repaired and checked by reading, not compiled, not run on hardware
//
#include <hip/hip_runtime.h>

// ---------------------------------------------------------------------------
// DA-RNN forward for MI455X (gfx1250, wave32, WMMA bf16 -> f32 accumulate)
// B=128, T=200, K=512, HE=256, HD=256
// All WMMA operands are pre-converted to zero-padded bf16 so fragment loads
// are unconditional b128 pairs; biases are zero-padded so epilogues are
// branch-free; the tanh-attention loop streams Ux with float4 (b128) loads.
// ---------------------------------------------------------------------------

typedef __attribute__((ext_vector_type(16))) __bf16 v16bf;
typedef __attribute__((ext_vector_type(8)))  __bf16 v8bf;
typedef __attribute__((ext_vector_type(8)))  float  v8f;

#define Bn 128
#define Tn 200
#define Kn 512
#define HEn 256
#define HDn 256
#define TP 208            // T padded for f32 Ux rows / bias vectors (mult 16)
#define TPK 224           // T padded for bf16 K-dim (mult of 32)

__device__ __forceinline__ __bf16 to_bf16(float f) { return (__bf16)f; }
__device__ __forceinline__ float sigmoidf(float x) { return 1.f / (1.f + __expf(-x)); }

__device__ __forceinline__ v16bf cat8(v8bf lo, v8bf hi) {
  return __builtin_shufflevector(lo, hi, 0, 1, 2, 3, 4, 5, 6, 7,
                                          8, 9, 10, 11, 12, 13, 14, 15);
}

// A fragment (16x32 bf16, contiguous k order as in fragB): lane(row=lane&15, half=lane>>4)
// elements 0..7  <- K = 16*half  .. +8   (16B contiguous)
// elements 8..15 <- K = 16*half+8 .. +8 (16B contiguous)
__device__ __forceinline__ v16bf fragA(const __bf16* base, int ld, int lane) {
  const int row = lane & 15, half = lane >> 4;
  const __bf16* p = base + (size_t)row * ld;
  v8bf lo = *(const v8bf*)(p + 16 * half);
  v8bf hi = *(const v8bf*)(p + 16 * half + 8);
  return cat8(lo, hi);
}
// B fragment (32x16 bf16): lane(n=lane&15, half) holds K = 16*half .. +16
// as 32 contiguous bytes of row n (N-major staging).
__device__ __forceinline__ v16bf fragB(const __bf16* base, int ld, int lane) {
  const int n = lane & 15, half = lane >> 4;
  const __bf16* p = base + (size_t)n * ld + 16 * half;
  v8bf lo = *(const v8bf*)(p);
  v8bf hi = *(const v8bf*)(p + 8);
  return cat8(lo, hi);
}

// ---------------------------------------------------------------------------
// Prep kernels
// ---------------------------------------------------------------------------
__global__ __launch_bounds__(256) void pad_convert(const float* __restrict__ src,
                                                   __bf16* __restrict__ dst,
                                                   int R, int C, int Rp, int Cp) {
  int i = blockIdx.x * 256 + threadIdx.x;
  if (i >= Rp * Cp) return;
  int r = i / Cp, c = i - r * Cp;
  dst[i] = (r < R && c < C) ? to_bf16(src[r * C + c]) : (__bf16)0.f;
}

__global__ __launch_bounds__(256) void pad_vec(const float* __restrict__ src,
                                               float* __restrict__ dst, int R, int Rp) {
  int i = blockIdx.x * 256 + threadIdx.x;
  if (i < Rp) dst[i] = (i < R) ? src[i] : 0.f;
}

// x[b][t][k] f32 -> xT[b][k][TPK] bf16 (LDS-tiled transpose, zero pad)
__global__ __launch_bounds__(256) void x_transpose(const float* __restrict__ x,
                                                   __bf16* __restrict__ xT) {
  __shared__ float tile[32][65];
  const int t0 = blockIdx.x * 32;
  const int k0 = blockIdx.y * 64;
  const int b = blockIdx.z;
  const int tid = threadIdx.x;
  for (int i = tid; i < 32 * 64; i += 256) {
    int tl = i >> 6, kl = i & 63;
    int t = t0 + tl;
    tile[tl][kl] = (t < Tn) ? x[((size_t)b * Tn + t) * Kn + k0 + kl] : 0.f;
  }
  __syncthreads();
  for (int i = tid; i < 64 * 32; i += 256) {
    int kl = i >> 5, tl = i & 31;
    xT[((size_t)b * Kn + k0 + kl) * TPK + t0 + tl] = to_bf16(tile[tl][kl]);
  }
}

// ---------------------------------------------------------------------------
// Kernel 1: Ux[b,k,s] = sum_t x[b,t,k] * Ue_w[s,t] + Ue_b[s]
// A = xT_bf[b][k][t], B = Uew_bf[s][t]; wave -> 16x16 tile, 7 K-chunks of 32.
// grid = (13 s-tiles, 4 k-macros, 128 b), block 256 (8 waves)
// ---------------------------------------------------------------------------
__global__ __launch_bounds__(256) void ux_gemm(const __bf16* __restrict__ xT,
                                               const __bf16* __restrict__ Uew,
                                               const float* __restrict__ Ueb_pad,
                                               float* __restrict__ Ux) {
  const int tid = threadIdx.x, lane = tid & 31, wave = tid >> 5;
  const int s0 = blockIdx.x * 16;
  const int k0 = blockIdx.y * 128 + wave * 16;
  const int b = blockIdx.z;
  const __bf16* aBase = xT + ((size_t)b * Kn + k0) * TPK;
  const __bf16* bBase = Uew + (size_t)s0 * TPK;
  v8f acc = {};
  for (int t0 = 0; t0 < TPK; t0 += 32) {
    if (t0 + 32 < TPK) {
      __builtin_prefetch(aBase + t0 + 32, 0, 3);
      __builtin_prefetch(bBase + t0 + 32, 0, 3);
    }
    v16bf a = fragA(aBase + t0, TPK, lane);
    v16bf bm = fragB(bBase + t0, TPK, lane);
    acc = __builtin_amdgcn_wmma_f32_16x16x32_bf16(false, a, false, bm,
                                                  (short)0, acc, false, false);
  }
  const int half = lane >> 4;
  const int ss = s0 + (lane & 15);
  const float bias = Ueb_pad[ss];                 // branch-free (zero padded)
  #pragma unroll
  for (int r = 0; r < 8; ++r) {
    int kk = k0 + r + 8 * half;
    Ux[((size_t)b * Kn + kk) * TP + ss] = acc[r] + bias;
  }
}

// ---------------------------------------------------------------------------
// Kernel 2: encoder scan over T=200. 8 blocks x 256 threads, 16 batch rows
// per block; h/c f32 state + bf16 A-operand mirror live in LDS.
// sh_ab cols: [0,256)=h  [256,512)=c  [512,1024)=x_tilde   (bf16)
// ---------------------------------------------------------------------------
#define BPW 16
__global__ __launch_bounds__(256) void encoder_scan(
    const float* __restrict__ x, const __bf16* __restrict__ Wew,
    const float* __restrict__ Web_pad, const float* __restrict__ ve_w,
    const float* __restrict__ ve_b, const __bf16* __restrict__ wih,
    const __bf16* __restrict__ whh, const float* __restrict__ enc_bih,
    const float* __restrict__ enc_bhh, const float* __restrict__ Ux,
    float* __restrict__ h_enc, __bf16* __restrict__ h_enc_bf) {
  __shared__ __attribute__((aligned(16))) float sh_h[BPW][HEn];       // 16 KB
  __shared__ __attribute__((aligned(16))) float sh_c[BPW][HEn];       // 16 KB
  __shared__ __attribute__((aligned(16))) float sh_w[BPW][TP];        // 13 KB
  __shared__ __attribute__((aligned(16))) float sh_g[BPW * 4 * HEn];  // 64 KB
  __shared__ __attribute__((aligned(16))) __bf16 sh_ab[BPW][1024];    // 32 KB
  __shared__ __attribute__((aligned(16))) float sh_vew[TP];

  const int b0 = blockIdx.x * BPW;
  const int tid = threadIdx.x;
  const int wave = tid >> 5, lane = tid & 31;
  const int half = lane >> 4;

  for (int i = tid; i < BPW * HEn; i += 256) {
    (&sh_h[0][0])[i] = 0.f;
    (&sh_c[0][0])[i] = 0.f;
  }
  for (int i = tid; i < BPW * 1024; i += 256) (&sh_ab[0][0])[i] = (__bf16)0.f;
  for (int i = tid; i < TP; i += 256) sh_vew[i] = (i < Tn) ? ve_w[i] : 0.f;
  __syncthreads();

  const __bf16* abBase = &sh_ab[0][0];

  for (int t = 0; t < Tn; ++t) {
    // --- Phase 1: w[b,s] = hs @ We_w^T + We_b  (A: LDS bf16, B: global bf16)
    for (int nt = wave; nt < 13; nt += 8) {            // wave-uniform
      const int s0n = nt * 16;
      v8f acc = {};
      for (int j0 = 0; j0 < 2 * HEn; j0 += 32) {
        v16bf a = fragA(abBase + j0, 1024, lane);
        v16bf bm = fragB(Wew + (size_t)s0n * (2 * HEn) + j0, 2 * HEn, lane);
        acc = __builtin_amdgcn_wmma_f32_16x16x32_bf16(false, a, false, bm,
                                                      (short)0, acc, false, false);
      }
      const int s = s0n + (lane & 15);
      const float bias = Web_pad[s];                   // branch-free
      #pragma unroll
      for (int r = 0; r < 8; ++r) sh_w[r + 8 * half][s] = acc[r] + bias;
    }
    __syncthreads();

    // --- Phase 2: e[b,k] = sum_s tanh(w[b,s]+Ux[b,k,s]) * ve_w[s] + ve_b
    //     (b128 streaming of Ux; e stored in sh_g scratch)
    for (int idx = tid; idx < BPW * Kn; idx += 256) {
      const int bb = idx >> 9, k = idx & (Kn - 1);
      const float* uxp = Ux + ((size_t)(b0 + bb) * Kn + k) * TP;
      float sum = 0.f;
      for (int s = 0; s < Tn; s += 4) {
        float4 u = *(const float4*)(uxp + s);
        float4 w4 = *(const float4*)(&sh_w[bb][s]);
        float4 v4 = *(const float4*)(&sh_vew[s]);
        sum += tanhf(w4.x + u.x) * v4.x;
        sum += tanhf(w4.y + u.y) * v4.y;
        sum += tanhf(w4.z + u.z) * v4.z;
        sum += tanhf(w4.w + u.w) * v4.w;
      }
      sh_g[idx] = sum + ve_b[0];
    }
    __syncthreads();

    // --- Phase 3: softmax over k; write x_tilde as bf16 into sh_ab[.,512+k]
    for (int bb = wave; bb < BPW; bb += 8) {           // wave-uniform
      float* e = sh_g + bb * Kn;
      float m = -3.4e38f;
      for (int k = lane; k < Kn; k += 32) m = fmaxf(m, e[k]);
      for (int o = 16; o; o >>= 1) m = fmaxf(m, __shfl_xor(m, o, 32));
      float sum = 0.f;
      for (int k = lane; k < Kn; k += 32) {
        float v = __expf(e[k] - m);
        e[k] = v;
        sum += v;
      }
      for (int o = 16; o; o >>= 1) sum += __shfl_xor(sum, o, 32);
      const float inv = 1.f / sum;
      const float* xrow = x + ((size_t)(b0 + bb) * Tn + t) * Kn;
      for (int k = lane; k < Kn; k += 32)
        sh_ab[bb][512 + k] = to_bf16(e[k] * inv * xrow[k]);
    }
    __syncthreads();

    // --- Phase 4: gates = x_tilde@wih^T + h@whh^T + biases (WMMA)
    for (int nt = wave; nt < 64; nt += 8) {            // wave-uniform
      const int n0 = nt * 16;
      v8f acc = {};
      for (int j0 = 0; j0 < Kn; j0 += 32) {            // A = x_tilde
        v16bf a = fragA(abBase + 512 + j0, 1024, lane);
        v16bf bm = fragB(wih + (size_t)n0 * Kn + j0, Kn, lane);
        acc = __builtin_amdgcn_wmma_f32_16x16x32_bf16(false, a, false, bm,
                                                      (short)0, acc, false, false);
      }
      for (int j0 = 0; j0 < HEn; j0 += 32) {           // A = h
        v16bf a = fragA(abBase + j0, 1024, lane);
        v16bf bm = fragB(whh + (size_t)n0 * HEn + j0, HEn, lane);
        acc = __builtin_amdgcn_wmma_f32_16x16x32_bf16(false, a, false, bm,
                                                      (short)0, acc, false, false);
      }
      const int n = n0 + (lane & 15);
      const float bias = enc_bih[n] + enc_bhh[n];
      #pragma unroll
      for (int r = 0; r < 8; ++r)
        sh_g[(r + 8 * half) * (4 * HEn) + n] = acc[r] + bias;
    }
    __syncthreads();

    // --- Phase 5: LSTM elementwise update; refresh bf16 mirror; emit h_enc
    for (int idx = tid; idx < BPW * HEn; idx += 256) {
      const int bb = idx >> 8, n = idx & (HEn - 1);
      const float* g = sh_g + bb * (4 * HEn);
      float ig = sigmoidf(g[n]);
      float fg = sigmoidf(g[HEn + n]);
      float gg = tanhf(g[2 * HEn + n]);
      float og = sigmoidf(g[3 * HEn + n]);
      float cn = fg * sh_c[bb][n] + ig * gg;
      float hn = og * tanhf(cn);
      sh_c[bb][n] = cn;
      sh_h[bb][n] = hn;
      sh_ab[bb][n] = to_bf16(hn);
      sh_ab[bb][HEn + n] = to_bf16(cn);
      size_t o = ((size_t)(b0 + bb) * Tn + t) * HEn + n;
      h_enc[o] = hn;
      h_enc_bf[o] = to_bf16(hn);
    }
    __syncthreads();
  }
}

// ---------------------------------------------------------------------------
// Kernel 3: l[row] = sum_h tanh(Wd_b[h]+Ud_b[h]+ h_enc[row,:]·Ud_w[h,:])*vd_w[h]
//           + vd_b,   rows = b*Tn+tt.  16 rows/block, WMMA 256x256 GEMM.
// ---------------------------------------------------------------------------
__global__ __launch_bounds__(256) void dec_attn(const __bf16* __restrict__ hencb,
                                                const __bf16* __restrict__ Udw,
                                                const float* __restrict__ Wd_b,
                                                const float* __restrict__ Ud_b,
                                                const float* __restrict__ vd_w,
                                                const float* __restrict__ vd_b,
                                                float* __restrict__ l_ws) {
  __shared__ __attribute__((aligned(16))) float sh_q[16][HEn];
  const int tid = threadIdx.x, lane = tid & 31, wave = tid >> 5;
  const int r0 = blockIdx.x * 16;
  const int half = lane >> 4;

  for (int nt = wave; nt < 16; nt += 8) {              // wave-uniform, 2 iters
    const int n0 = nt * 16;
    v8f acc = {};
    for (int j0 = 0; j0 < HEn; j0 += 32) {
      v16bf a = fragA(hencb + (size_t)r0 * HEn + j0, HEn, lane);
      v16bf bm = fragB(Udw + (size_t)n0 * HEn + j0, HEn, lane);
      acc = __builtin_amdgcn_wmma_f32_16x16x32_bf16(false, a, false, bm,
                                                    (short)0, acc, false, false);
    }
    const int n = n0 + (lane & 15);
    const float bias = Wd_b[n] + Ud_b[n];
    #pragma unroll
    for (int r = 0; r < 8; ++r) sh_q[r + 8 * half][n] = acc[r] + bias;
  }
  __syncthreads();
  for (int rr = wave; rr < 16; rr += 8) {
    float sum = 0.f;
    for (int h = lane; h < HEn; h += 32) sum += tanhf(sh_q[rr][h]) * vd_w[h];
    for (int o = 16; o; o >>= 1) sum += __shfl_xor(sum, o, 32);
    if (lane == 0) l_ws[r0 + rr] = sum + vd_b[0];
  }
}

// ---------------------------------------------------------------------------
// Kernel 4: per-batch softmax over T, context c_t, y_tilde, decoder LSTM
// (d=s=0), output head. One block (256 thr) per batch row.
// ---------------------------------------------------------------------------
__global__ __launch_bounds__(256) void dec_head(
    const float* __restrict__ henc, const float* __restrict__ l_ws,
    const float* __restrict__ y, const float* __restrict__ wt_w,
    const float* __restrict__ wt_b, const float* __restrict__ dec_wih,
    const float* __restrict__ dec_bih, const float* __restrict__ dec_bhh,
    const float* __restrict__ Wy_w, const float* __restrict__ Wy_b,
    const float* __restrict__ vy_w, const float* __restrict__ vy_b,
    float* __restrict__ out) {
  __shared__ float sh_beta[256];
  __shared__ float sh_ct[HEn];
  __shared__ __attribute__((aligned(16))) float sh_cat[2 * HDn];
  __shared__ float red[8];
  __shared__ float sc[2];
  const int b = blockIdx.x, tid = threadIdx.x;
  const int lane = tid & 31, wave = tid >> 5;

  float v = (tid < Tn) ? l_ws[b * Tn + tid] : -3.4e38f;
  float m = v;
  for (int o = 16; o; o >>= 1) m = fmaxf(m, __shfl_xor(m, o, 32));
  if (lane == 0) red[wave] = m;
  __syncthreads();
  if (tid == 0) {
    float mm = red[0];
    for (int i = 1; i < 8; ++i) mm = fmaxf(mm, red[i]);
    sc[0] = mm;
  }
  __syncthreads();
  float e = (tid < Tn) ? __expf(v - sc[0]) : 0.f;
  float ssum = e;
  for (int o = 16; o; o >>= 1) ssum += __shfl_xor(ssum, o, 32);
  if (lane == 0) red[wave] = ssum;
  __syncthreads();
  if (tid == 0) {
    float t = 0.f;
    for (int i = 0; i < 8; ++i) t += red[i];
    sc[1] = t;
  }
  __syncthreads();
  sh_beta[tid] = e / sc[1];
  __syncthreads();

  float ct = 0.f;
  for (int tt = 0; tt < Tn; ++tt)
    ct += sh_beta[tt] * henc[((size_t)b * Tn + tt) * HEn + tid];
  sh_ct[tid] = ct;
  __syncthreads();

  float part = ct * wt_w[tid];
  for (int o = 16; o; o >>= 1) part += __shfl_xor(part, o, 32);
  if (lane == 0) red[wave] = part;
  __syncthreads();
  if (tid == 0) {
    float t = 0.f;
    for (int i = 0; i < 8; ++i) t += red[i];
    sc[0] = t + y[(size_t)b * Tn] * wt_w[HEn] + wt_b[0];
  }
  __syncthreads();
  const float y_tilde = sc[0];

  const int n = tid;
  float gi = y_tilde * dec_wih[n] + dec_bih[n] + dec_bhh[n];
  float gg = y_tilde * dec_wih[2 * HDn + n] + dec_bih[2 * HDn + n] + dec_bhh[2 * HDn + n];
  float go = y_tilde * dec_wih[3 * HDn + n] + dec_bih[3 * HDn + n] + dec_bhh[3 * HDn + n];
  float cnew = sigmoidf(gi) * tanhf(gg);
  float d1 = sigmoidf(go) * tanhf(cnew);
  sh_cat[n] = d1;
  sh_cat[HDn + n] = sh_ct[n];
  __syncthreads();

  float z = Wy_b[tid];
  for (int j = 0; j < 2 * HDn; j += 4) {
    float4 c4 = *(const float4*)(&sh_cat[j]);
    const float* wr = Wy_w + (size_t)tid * (2 * HDn) + j;
    z += c4.x * wr[0] + c4.y * wr[1] + c4.z * wr[2] + c4.w * wr[3];
  }
  float p = z * vy_w[tid];
  for (int o = 16; o; o >>= 1) p += __shfl_xor(p, o, 32);
  if (lane == 0) red[wave] = p;
  __syncthreads();
  if (tid == 0) {
    float t = 0.f;
    for (int i = 0; i < 8; ++i) t += red[i];
    out[b] = t + vy_b[0];
  }
}

// ---------------------------------------------------------------------------
extern "C" void kernel_launch(void* const* d_in, const int* in_sizes, int n_in,
                              void* d_out, int out_size, void* d_ws, size_t ws_size,
                              hipStream_t stream) {
  (void)in_sizes; (void)n_in; (void)out_size; (void)ws_size;
  const float* x       = (const float*)d_in[0];
  const float* y       = (const float*)d_in[1];
  const float* We_w    = (const float*)d_in[2];
  const float* We_b    = (const float*)d_in[3];
  const float* Ue_w    = (const float*)d_in[4];
  const float* Ue_b    = (const float*)d_in[5];
  const float* ve_w    = (const float*)d_in[6];
  const float* ve_b    = (const float*)d_in[7];
  const float* enc_wih = (const float*)d_in[8];
  const float* enc_whh = (const float*)d_in[9];
  const float* enc_bih = (const float*)d_in[10];
  const float* enc_bhh = (const float*)d_in[11];
  const float* Wd_b    = (const float*)d_in[13];
  const float* Ud_w    = (const float*)d_in[14];
  const float* Ud_b    = (const float*)d_in[15];
  const float* vd_w    = (const float*)d_in[16];
  const float* vd_b    = (const float*)d_in[17];
  const float* wt_w    = (const float*)d_in[18];
  const float* wt_b    = (const float*)d_in[19];
  const float* dec_wih = (const float*)d_in[20];
  const float* dec_bih = (const float*)d_in[22];
  const float* dec_bhh = (const float*)d_in[23];
  const float* Wy_w    = (const float*)d_in[24];
  const float* Wy_b    = (const float*)d_in[25];
  const float* vy_w    = (const float*)d_in[26];
  const float* vy_b    = (const float*)d_in[27];
  float* out = (float*)d_out;

  // ---- workspace layout ----
  float* ws = (float*)d_ws;
  float* Ux      = ws;                                // 128*512*208 f32
  float* henc    = Ux + (size_t)Bn * Kn * TP;         // 128*200*256 f32
  float* l_ws    = henc + (size_t)Bn * Tn * HEn;      // 25600 f32
  float* Ueb_pad = l_ws + (size_t)Bn * Tn;            // 208 f32
  float* Web_pad = Ueb_pad + TP;                      // 208 f32
  __bf16* bfb = (__bf16*)(Web_pad + TP);
  __bf16* xT_bf   = bfb;                              // 128*512*224
  __bf16* Uew_bf  = xT_bf + (size_t)Bn * Kn * TPK;    // 208*224
  __bf16* Wew_bf  = Uew_bf + (size_t)TP * TPK;        // 208*512
  __bf16* wih_bf  = Wew_bf + (size_t)TP * (2 * HEn);  // 1024*512
  __bf16* whh_bf  = wih_bf + (size_t)(4 * HEn) * Kn;  // 1024*256
  __bf16* Udw_bf  = whh_bf + (size_t)(4 * HEn) * HEn; // 256*256
  __bf16* henc_bf = Udw_bf + (size_t)HEn * HEn;       // 128*200*256

  // ---- prep: bf16 conversions / padding / transpose ----
  pad_convert<<<(TP * TPK + 255) / 256, 256, 0, stream>>>(Ue_w, Uew_bf, Tn, Tn, TP, TPK);
  pad_convert<<<(TP * 2 * HEn + 255) / 256, 256, 0, stream>>>(We_w, Wew_bf, Tn, 2 * HEn, TP, 2 * HEn);
  pad_convert<<<(4 * HEn * Kn + 255) / 256, 256, 0, stream>>>(enc_wih, wih_bf, 4 * HEn, Kn, 4 * HEn, Kn);
  pad_convert<<<(4 * HEn * HEn + 255) / 256, 256, 0, stream>>>(enc_whh, whh_bf, 4 * HEn, HEn, 4 * HEn, HEn);
  pad_convert<<<(HEn * HEn + 255) / 256, 256, 0, stream>>>(Ud_w, Udw_bf, HEn, HEn, HEn, HEn);
  pad_vec<<<1, 256, 0, stream>>>(Ue_b, Ueb_pad, Tn, TP);
  pad_vec<<<1, 256, 0, stream>>>(We_b, Web_pad, Tn, TP);
  x_transpose<<<dim3(TPK / 32, Kn / 64, Bn), 256, 0, stream>>>(x, xT_bf);

  // ---- main pipeline ----
  ux_gemm<<<dim3(13, Kn / 128, Bn), 256, 0, stream>>>(xT_bf, Uew_bf, Ueb_pad, Ux);
  encoder_scan<<<Bn / BPW, 256, 0, stream>>>(x, Wew_bf, Web_pad, ve_w, ve_b,
                                             wih_bf, whh_bf, enc_bih, enc_bhh,
                                             Ux, henc, henc_bf);
  dec_attn<<<(Bn * Tn) / 16, 256, 0, stream>>>(henc_bf, Udw_bf, Wd_b, Ud_b,
                                               vd_w, vd_b, l_ws);
  dec_head<<<Bn, 256, 0, stream>>>(henc, l_ws, y, wt_w, wt_b,
                                   dec_wih, dec_bih, dec_bhh,
                                   Wy_w, Wy_b, vy_w, vy_b, out);
}
